// PAINNBlock_49641232007324
// MI455X (gfx1250) — matrix operand order repaired and checked by reading, not compiled, not run on hardware
//
#include <hip/hip_runtime.h>
#include <hip/hip_bf16.h>
#include <math.h>

// ---------------------------------------------------------------------------
// PaiNN block on MI455X (gfx1250, wave32, WMMA).
// Fully fused per-phase kernels: weights live in LDS pre-swizzled into
// B-fragment order (so a B-frag load is 2x ds_load_b128), each wave owns a
// 16-edge tile and runs the whole GEMM chain with v_wmma_f32_16x16x32_f16,
// scattering results with global f32 atomics. x (20MB) is L2-resident.
// ---------------------------------------------------------------------------

typedef __attribute__((ext_vector_type(16))) _Float16 v16h;
typedef __attribute__((ext_vector_type(8)))  _Float16 v8h;
typedef __attribute__((ext_vector_type(8)))  float    v8f;
typedef _Float16 half_t;

#define N_NODES 10000
#define N_EDGES 160000
#define R_CUT   1.4415f
#define PI_F    3.14159265358979f

// ---- WMMA fragment helpers (layouts per CDNA5 ISA 7.12.2) -----------------

__device__ __forceinline__ v8f wmma32(v16h a, v16h b, v8f c) {
  return __builtin_amdgcn_wmma_f32_16x16x32_f16(false, a, false, b, (short)0, c,
                                                false, false);
}

__device__ __forceinline__ v16h cat8(v8h lo, v8h hi) {
  return __builtin_shufflevector(lo, hi, 0, 1, 2, 3, 4, 5, 6, 7, 8, 9, 10, 11,
                                 12, 13, 14, 15);
}

// A fragment (16x32): per lane two contiguous 8-half runs at kb+h*8, kb+16+h*8.
__device__ __forceinline__ v16h load_a_lds(const half_t* __restrict__ row,
                                           int halfId, int kb) {
  const v8h* p0 = (const v8h*)(row + kb + halfId * 8);
  const v8h* p1 = (const v8h*)(row + kb + 16 + halfId * 8);
  return cat8(*p0, *p1);
}

// A fragment from an f32 row (gather + convert), 4x float4 loads.
__device__ __forceinline__ v16h load_a_f32(const float* __restrict__ row,
                                           int halfId, int kb) {
  const float4* q0 = (const float4*)(row + kb + halfId * 8);
  const float4* q1 = (const float4*)(row + kb + 16 + halfId * 8);
  float4 a0 = q0[0], a1 = q0[1], b0 = q1[0], b1 = q1[1];
  v16h a;
  a[0] = (half_t)a0.x;  a[1] = (half_t)a0.y;  a[2] = (half_t)a0.z;  a[3] = (half_t)a0.w;
  a[4] = (half_t)a1.x;  a[5] = (half_t)a1.y;  a[6] = (half_t)a1.z;  a[7] = (half_t)a1.w;
  a[8] = (half_t)b0.x;  a[9] = (half_t)b0.y;  a[10] = (half_t)b0.z; a[11] = (half_t)b0.w;
  a[12] = (half_t)b1.x; a[13] = (half_t)b1.y; a[14] = (half_t)b1.z; a[15] = (half_t)b1.w;
  return a;
}

// B fragment from pre-swizzled weights: tile = 512 contiguous halves,
// lane's 16 halves contiguous -> 2x ds_load_b128.
__device__ __forceinline__ v16h load_b_swz(const half_t* __restrict__ Wswz,
                                           int tileIdx, int lane) {
  const v8h* p = (const v8h*)(Wswz + (size_t)tileIdx * 512 + lane * 16);
  return cat8(p[0], p[1]);
}

__device__ __forceinline__ v8f splat8(float v) {
  v8f c;
#pragma unroll
  for (int i = 0; i < 8; ++i) c[i] = v;
  return c;
}

__device__ __forceinline__ float silu(float v) {
  return v / (1.0f + __expf(-v));
}

__device__ __forceinline__ float geth(v8h u0, v8h u1, v8h u2, int t) {
  return (t < 8) ? (float)u0[t] : (t < 16) ? (float)u1[t - 8] : (float)u2[t - 16];
}

// ---- weight conversion (f32 row-major -> f16 B-fragment-swizzled) ---------
// dest element i: tile = i/512 (tile order: nt-major, kt-minor), lane = (i/16)%32,
// e = i%16 ; source element W[kt*32 + (e>>3)*16 + (lane>>4)*8 + (e&7)][nt*16 + (lane&15)].

__global__ void cvt_swz(const float* __restrict__ s, half_t* __restrict__ d,
                        int K, int N, int Ksrc, int total) {
  int i = blockIdx.x * 256 + threadIdx.x;
  if (i >= total) return;
  int tile = i >> 9;
  int rem  = i & 511;
  int lane = rem >> 4;
  int e    = rem & 15;
  int numKT = K >> 5;
  int nt = tile / numKT, kt = tile - nt * numKT;
  int k = kt * 32 + ((e >> 3) << 4) + ((lane >> 4) << 3) + (e & 7);
  int n = nt * 16 + (lane & 15);
  d[i] = (k < Ksrc) ? (half_t)s[k * N + n] : (half_t)0.0f;
}

// Wu2 (128x384) -> swizzled (128x256): cols 0..127 = s1, 128..255 = s3 (src col+128)
__global__ void cvt_swz_wu2(const float* __restrict__ s, half_t* __restrict__ d) {
  int i = blockIdx.x * 256 + threadIdx.x;
  if (i >= 128 * 256) return;
  int tile = i >> 9;
  int rem  = i & 511;
  int lane = rem >> 4;
  int e    = rem & 15;
  int nt = tile >> 2, kt = tile & 3;  // numKT = 4
  int k = kt * 32 + ((e >> 3) << 4) + ((lane >> 4) << 3) + (e & 7);
  int n = nt * 16 + (lane & 15);
  int cs = (n < 128) ? n : (n + 128);
  d[i] = (half_t)s[k * 384 + cs];
}

__global__ void fcopy(const float* __restrict__ s, float* __restrict__ d, int n) {
  int i = blockIdx.x * 256 + threadIdx.x;
  if (i < n) d[i] = s[i];
}

// ---- Phase 1 ---------------------------------------------------------------
#define P1_WAVES 8
#define P1_WOFF  155648   // bytes of weights in LDS
#define P1_WSTRIDE 4608

__global__ __launch_bounds__(256) void painn_phase1(
    const float* __restrict__ x, const int* __restrict__ ei,
    const float* __restrict__ ea1, const float* __restrict__ ea2,
    const float* __restrict__ b1, const float* __restrict__ b2,
    const float* __restrict__ b3, const half_t* __restrict__ wsW,
    float* __restrict__ xout) {
  extern __shared__ char smem[];
  {
    const uint4* s = (const uint4*)wsW;
    uint4* d = (uint4*)smem;
    for (int i = threadIdx.x; i < P1_WOFF / 16; i += 256) d[i] = s[i];
  }
  half_t* W1s = (half_t*)smem;            // swz 128x128 (numKT=4)
  half_t* W3s = W1s + 16384;              // swz 128x384 (numKT=4)
  half_t* W2s = W3s + 49152;              // swz 32x384  (numKT=1)

  int w      = threadIdx.x >> 5;
  int lane   = threadIdx.x & 31;
  int m16    = lane & 15;
  int halfId = (lane >> 4) & 1;

  char* pw   = smem + P1_WOFF + w * P1_WSTRIDE;
  half_t* H1 = (half_t*)pw;               // [16][128]
  int* srcs  = (int*)(pw + 4096);
  int* dsts  = (int*)(pw + 4160);
  float* dirs = (float*)(pw + 4224);      // [16][3]

  int tile = blockIdx.x * P1_WAVES + w;
  int e0   = tile * 16;
  if (lane < 16) {
    int e = e0 + lane;
    srcs[lane] = ei[e];
    dsts[lane] = ei[N_EDGES + e];
    dirs[lane * 3 + 0] = ea1[e * 3 + 0];
    dirs[lane * 3 + 1] = ea1[e * 3 + 1];
    dirs[lane * 3 + 2] = ea1[e * 3 + 2];
  }
  __syncthreads();

  // --- stage 1: H1 = silu(s_j @ W1 + b1) ---
  const float* arow = x + (size_t)srcs[m16] * 512 + 384;
  v16h aj0 = load_a_f32(arow, halfId, 0);
  v16h aj1 = load_a_f32(arow, halfId, 32);
  v16h aj2 = load_a_f32(arow, halfId, 64);
  v16h aj3 = load_a_f32(arow, halfId, 96);
#pragma unroll
  for (int nt = 0; nt < 8; ++nt) {
    v8f c = splat8(b1[nt * 16 + m16]);
    c = wmma32(aj0, load_b_swz(W1s, nt * 4 + 0, lane), c);
    c = wmma32(aj1, load_b_swz(W1s, nt * 4 + 1, lane), c);
    c = wmma32(aj2, load_b_swz(W1s, nt * 4 + 2, lane), c);
    c = wmma32(aj3, load_b_swz(W1s, nt * 4 + 3, lane), c);
#pragma unroll
    for (int i = 0; i < 8; ++i)
      H1[(i + 8 * halfId) * 128 + nt * 16 + m16] = (half_t)silu(c[i]);
  }
  __syncthreads();

  v16h ah0 = load_a_lds(H1 + m16 * 128, halfId, 0);
  v16h ah1 = load_a_lds(H1 + m16 * 128, halfId, 32);
  v16h ah2 = load_a_lds(H1 + m16 * 128, halfId, 64);
  v16h ah3 = load_a_lds(H1 + m16 * 128, halfId, 96);

  // --- RBF A fragment (K=32, 20 valid) ---
  float r  = ea2[e0 + m16];
  float co = (r < R_CUT) ? 0.5f * (__cosf(PI_F * r / R_CUT) + 1.0f) : 0.0f;
  float sc = sqrtf(2.0f / R_CUT) * co / r;
  float w0 = PI_F * r / R_CUT;
  v16h arbf;
#pragma unroll
  for (int p = 0; p < 8; ++p) {
    int k = (p < 4) ? (halfId * 8 + 2 * p) : (16 + halfId * 8 + 2 * (p - 4));
    float v0 = (k < 20)     ? sc * __sinf((float)(k + 1) * w0) : 0.0f;
    float v1 = (k + 1 < 20) ? sc * __sinf((float)(k + 2) * w0) : 0.0f;
    arbf[2 * p]     = (half_t)v0;
    arbf[2 * p + 1] = (half_t)v1;
  }

  // --- s1/s3 paired tiles with fused vector scatter ---
#pragma unroll
  for (int nt = 0; nt < 8; ++nt) {
    int n1 = nt * 16 + m16, n3 = n1 + 256;
    v8f c1 = splat8(b3[n1]), c3 = splat8(b3[n3]);
    v8f e1 = splat8(b2[n1]), e3 = splat8(b2[n3]);
#pragma unroll
    for (int q = 0; q < 4; ++q) {
      c1 = wmma32(q == 0 ? ah0 : q == 1 ? ah1 : q == 2 ? ah2 : ah3,
                  load_b_swz(W3s, nt * 4 + q, lane), c1);
      c3 = wmma32(q == 0 ? ah0 : q == 1 ? ah1 : q == 2 ? ah2 : ah3,
                  load_b_swz(W3s, (nt + 16) * 4 + q, lane), c3);
    }
    e1 = wmma32(arbf, load_b_swz(W2s, nt, lane), e1);
    e3 = wmma32(arbf, load_b_swz(W2s, nt + 16, lane), e3);
#pragma unroll
    for (int i = 0; i < 8; ++i) {
      int mrow = i + 8 * halfId;
      float s1 = c1[i] * e1[i];
      float s3 = c3[i] * e3[i];
      const float* vrow = x + (size_t)srcs[mrow] * 512;
      float* orow = xout + (size_t)dsts[mrow] * 512;
      int j = nt * 16 + m16;
#pragma unroll
      for (int k = 0; k < 3; ++k)
        atomicAdd(&orow[3 * j + k],
                  s1 * vrow[3 * j + k] + s3 * dirs[mrow * 3 + k]);
    }
  }
  // --- s2 tiles ---
#pragma unroll
  for (int nt = 8; nt < 16; ++nt) {
    int n2 = nt * 16 + m16;
    v8f c2 = splat8(b3[n2]);
    v8f e2 = splat8(b2[n2]);
    c2 = wmma32(ah0, load_b_swz(W3s, nt * 4 + 0, lane), c2);
    c2 = wmma32(ah1, load_b_swz(W3s, nt * 4 + 1, lane), c2);
    c2 = wmma32(ah2, load_b_swz(W3s, nt * 4 + 2, lane), c2);
    c2 = wmma32(ah3, load_b_swz(W3s, nt * 4 + 3, lane), c2);
    e2 = wmma32(arbf, load_b_swz(W2s, nt, lane), e2);
#pragma unroll
    for (int i = 0; i < 8; ++i) {
      int mrow = i + 8 * halfId;
      float* orow = xout + (size_t)dsts[mrow] * 512;
      atomicAdd(&orow[384 + (n2 - 128)], c2[i] * e2[i]);
    }
  }
}

// ---- Phase 2 ---------------------------------------------------------------
#define P2_WAVES 4
#define P2_WOFF  196608
#define P2_WSTRIDE 28928

__global__ __launch_bounds__(128) void painn_phase2(
    const float* __restrict__ x1, const int* __restrict__ ei,
    const float* __restrict__ bu1, const float* __restrict__ bu2,
    const half_t* __restrict__ wsW2, float* __restrict__ xout) {
  extern __shared__ char smem[];
  {
    const uint4* s = (const uint4*)wsW2;
    uint4* d = (uint4*)smem;
    for (int i = threadIdx.x; i < P2_WOFF / 16; i += 128) d[i] = s[i];
  }
  half_t* Us   = (half_t*)smem;        // swz 128x128 (numKT=4)
  half_t* Vs   = Us + 16384;           // swz 128x128 (numKT=4)
  half_t* Wu1s = Vs + 16384;           // swz 256x128 (numKT=8)
  half_t* Wu2s = Wu1s + 32768;         // swz 128x256 (numKT=4; nt 0..7=s1, 8..15=s3)

  int w      = threadIdx.x >> 5;
  int lane   = threadIdx.x & 31;
  int m16    = lane & 15;
  int halfId = (lane >> 4) & 1;

  char* pw   = smem + P2_WOFF + w * P2_WSTRIDE;
  half_t* VU = (half_t*)pw;            // [16][384] flat (3,128)
  half_t* VV = VU + 16 * 384;          // [16][384]
  half_t* H2 = VV + 16 * 384;          // [16][128]
  int* srcs  = (int*)(pw + 28672);
  int* dsts  = (int*)(pw + 28736);

  int tile = blockIdx.x * P2_WAVES + w;
  int e0   = tile * 16;
  if (lane < 16) {
    int e = e0 + lane;
    srcs[lane] = ei[e];
    dsts[lane] = ei[N_EDGES + e];
  }
  __syncthreads();

  const float* xsrow = x1 + (size_t)srcs[m16] * 512;

  // --- VU = (3x128 rows) @ U ---
#pragma unroll
  for (int rr = 0; rr < 3; ++rr) {
    v16h a0 = load_a_f32(xsrow + rr * 128, halfId, 0);
    v16h a1 = load_a_f32(xsrow + rr * 128, halfId, 32);
    v16h a2 = load_a_f32(xsrow + rr * 128, halfId, 64);
    v16h a3 = load_a_f32(xsrow + rr * 128, halfId, 96);
#pragma unroll
    for (int nt = 0; nt < 8; ++nt) {
      v8f c = splat8(0.0f);
      c = wmma32(a0, load_b_swz(Us, nt * 4 + 0, lane), c);
      c = wmma32(a1, load_b_swz(Us, nt * 4 + 1, lane), c);
      c = wmma32(a2, load_b_swz(Us, nt * 4 + 2, lane), c);
      c = wmma32(a3, load_b_swz(Us, nt * 4 + 3, lane), c);
#pragma unroll
      for (int i = 0; i < 8; ++i)
        VU[(i + 8 * halfId) * 384 + rr * 128 + nt * 16 + m16] = (half_t)c[i];
    }
  }
  __syncthreads();

  // --- VV = VU @ V ---
#pragma unroll
  for (int rr = 0; rr < 3; ++rr) {
    const half_t* vrow = VU + m16 * 384 + rr * 128;
    v16h a0 = load_a_lds(vrow, halfId, 0);
    v16h a1 = load_a_lds(vrow, halfId, 32);
    v16h a2 = load_a_lds(vrow, halfId, 64);
    v16h a3 = load_a_lds(vrow, halfId, 96);
#pragma unroll
    for (int nt = 0; nt < 8; ++nt) {
      v8f c = splat8(0.0f);
      c = wmma32(a0, load_b_swz(Vs, nt * 4 + 0, lane), c);
      c = wmma32(a1, load_b_swz(Vs, nt * 4 + 1, lane), c);
      c = wmma32(a2, load_b_swz(Vs, nt * 4 + 2, lane), c);
      c = wmma32(a3, load_b_swz(Vs, nt * 4 + 3, lane), c);
#pragma unroll
      for (int i = 0; i < 8; ++i)
        VV[(i + 8 * halfId) * 384 + rr * 128 + nt * 16 + m16] = (half_t)c[i];
    }
  }
  __syncthreads();

  // --- B2 fragments: K=256 ([norm(128), s(128)]) ---
  const half_t* vvrow = VV + m16 * 384;
  v16h bf[8];
#pragma unroll
  for (int f = 0; f < 4; ++f) {  // norm half: two 8-j runs, 3x b128 each
    v16h a;
#pragma unroll
    for (int run = 0; run < 2; ++run) {
      int j0 = f * 32 + run * 16 + halfId * 8;
      const v8h* pv = (const v8h*)(vvrow + 3 * j0);
      v8h u0 = pv[0], u1 = pv[1], u2 = pv[2];
#pragma unroll
      for (int jj = 0; jj < 8; ++jj) {
        float t0 = geth(u0, u1, u2, 3 * jj);
        float t1 = geth(u0, u1, u2, 3 * jj + 1);
        float t2 = geth(u0, u1, u2, 3 * jj + 2);
        a[run * 8 + jj] = (half_t)sqrtf(t0 * t0 + t1 * t1 + t2 * t2);
      }
    }
    bf[f] = a;
  }
#pragma unroll
  for (int f = 0; f < 4; ++f)  // s half = plain A fragments of x1[src][384:]
    bf[4 + f] = load_a_f32(xsrow + 384, halfId, f * 32);

  // --- H2 = silu(B2 @ Wu1 + bu1) ---
#pragma unroll
  for (int nt = 0; nt < 8; ++nt) {
    v8f c = splat8(bu1[nt * 16 + m16]);
#pragma unroll
    for (int q = 0; q < 8; ++q)
      c = wmma32(bf[q], load_b_swz(Wu1s, nt * 8 + q, lane), c);
#pragma unroll
    for (int i = 0; i < 8; ++i)
      H2[(i + 8 * halfId) * 128 + nt * 16 + m16] = (half_t)silu(c[i]);
  }
  __syncthreads();

  v16h h0 = load_a_lds(H2 + m16 * 128, halfId, 0);
  v16h h1 = load_a_lds(H2 + m16 * 128, halfId, 32);
  v16h h2 = load_a_lds(H2 + m16 * 128, halfId, 64);
  v16h h3 = load_a_lds(H2 + m16 * 128, halfId, 96);

  // --- H3 (s1,s3 only) + scatter ---
#pragma unroll
  for (int nt = 0; nt < 8; ++nt) {
    v8f c1 = splat8(bu2[nt * 16 + m16]);
    v8f c3 = splat8(bu2[256 + nt * 16 + m16]);
#pragma unroll
    for (int q = 0; q < 4; ++q) {
      v16h hq = q == 0 ? h0 : q == 1 ? h1 : q == 2 ? h2 : h3;
      c1 = wmma32(hq, load_b_swz(Wu2s, nt * 4 + q, lane), c1);
      c3 = wmma32(hq, load_b_swz(Wu2s, (nt + 8) * 4 + q, lane), c3);
    }
#pragma unroll
    for (int i = 0; i < 8; ++i) {
      int mrow = i + 8 * halfId;
      int j = nt * 16 + m16;
      const half_t* vu = VU + mrow * 384 + 3 * j;
      const half_t* vv = VV + mrow * 384 + 3 * j;
      float s1 = c1[i], s3 = c3[i];
      float v0 = s1 * (float)vu[0];
      float v1 = s1 * (float)vu[1];
      float v2 = s1 * (float)vu[2];
      float dot = v0 * (float)vv[0] + v1 * (float)vv[1] + v2 * (float)vv[2];
      float* orow = xout + (size_t)dsts[mrow] * 512;
      atomicAdd(&orow[3 * j + 0], v0);
      atomicAdd(&orow[3 * j + 1], v1);
      atomicAdd(&orow[3 * j + 2], v2);
      atomicAdd(&orow[384 + j], dot + s3);
    }
  }
}

// ---- launcher --------------------------------------------------------------
// ws layout (f16 elements, all swizzled):
//   W1s @0(16384) W3s @16384(49152) W2s @65536(12288)
//   Us @77824(16384) Vs @94208(16384) Wu1s @110592(32768) Wu2s @143360(32768)
//   x1buf (f32) @ byte 352256 (5,120,000 floats)
extern "C" void kernel_launch(void* const* d_in, const int* in_sizes, int n_in,
                              void* d_out, int out_size, void* d_ws,
                              size_t ws_size, hipStream_t stream) {
  const float* x    = (const float*)d_in[0];
  const int*   ei   = (const int*)d_in[1];
  const float* ea1  = (const float*)d_in[2];
  const float* ea2  = (const float*)d_in[3];
  const float* W1   = (const float*)d_in[4];
  const float* b1   = (const float*)d_in[5];
  const float* W2   = (const float*)d_in[6];
  const float* b2   = (const float*)d_in[7];
  const float* W3   = (const float*)d_in[8];
  const float* b3   = (const float*)d_in[9];
  const float* U    = (const float*)d_in[10];
  const float* V    = (const float*)d_in[11];
  const float* Wu1  = (const float*)d_in[12];
  const float* bu1  = (const float*)d_in[13];
  const float* Wu2  = (const float*)d_in[14];
  const float* bu2  = (const float*)d_in[15];

  half_t* wsW   = (half_t*)d_ws;
  float*  x1buf = (float*)((char*)d_ws + 352256);
  float*  xout  = (float*)d_out;
  (void)in_sizes; (void)n_in; (void)out_size; (void)ws_size;

  // f16 weight prep (B-fragment-swizzled)
  cvt_swz<<<64, 256, 0, stream>>>(W1, wsW, 128, 128, 128, 16384);
  cvt_swz<<<192, 256, 0, stream>>>(W3, wsW + 16384, 128, 384, 128, 49152);
  cvt_swz<<<48, 256, 0, stream>>>(W2, wsW + 65536, 32, 384, 20, 12288);
  cvt_swz<<<64, 256, 0, stream>>>(U, wsW + 77824, 128, 128, 128, 16384);
  cvt_swz<<<64, 256, 0, stream>>>(V, wsW + 94208, 128, 128, 128, 16384);
  cvt_swz<<<128, 256, 0, stream>>>(Wu1, wsW + 110592, 256, 128, 256, 32768);
  cvt_swz_wu2<<<128, 256, 0, stream>>>(Wu2, wsW + 143360);

  // xout = x
  fcopy<<<(N_NODES * 512 + 255) / 256, 256, 0, stream>>>(x, xout, N_NODES * 512);

  size_t p1_lds = P1_WOFF + P1_WAVES * P1_WSTRIDE;     // 192512 B
  size_t p2_lds = P2_WOFF + P2_WAVES * P2_WSTRIDE;     // 312320 B
  hipFuncSetAttribute(reinterpret_cast<const void*>(painn_phase1),
                      hipFuncAttributeMaxDynamicSharedMemorySize, (int)p1_lds);
  hipFuncSetAttribute(reinterpret_cast<const void*>(painn_phase2),
                      hipFuncAttributeMaxDynamicSharedMemorySize, (int)p2_lds);

  // phase 1: 10000 tiles of 16 edges, 8 waves per workgroup
  painn_phase1<<<N_EDGES / 16 / P1_WAVES, 256, p1_lds, stream>>>(
      x, ei, ea1, ea2, b1, b2, b3, wsW, xout);

  // snapshot post-phase-1 node state (phase 2 gathers it while scattering xout)
  fcopy<<<(N_NODES * 512 + 255) / 256, 256, 0, stream>>>(xout, x1buf,
                                                         N_NODES * 512);

  // phase 2: 10000 tiles, 4 waves per workgroup
  painn_phase2<<<N_EDGES / 16 / P2_WAVES, 128, p2_lds, stream>>>(
      x1buf, ei, bu1, bu2, wsW + 77824, xout);
}
